// CrossLevelAttention_74586402063030
// MI455X (gfx1250) — matrix operand order repaired and checked by reading, not compiled, not run on hardware
//
#include <hip/hip_runtime.h>
#include <hip/hip_bf16.h>
#include <math.h>

#define BATCH 8
#define SJN   3072
#define SSN   1024
#define DD    512

typedef __bf16 bf16;
typedef bf16  v16bf __attribute__((ext_vector_type(16)));
typedef bf16  v8bf  __attribute__((ext_vector_type(8)));
typedef bf16  v4bf  __attribute__((ext_vector_type(4)));
typedef float v8f   __attribute__((ext_vector_type(8)));

// ---- staged-chunk geometry ----
#define PADK 40                         // halfs per line (32 data + 8 pad) -> 80B line, bank-conflict free
#define CHUNK_HALFS (DD * PADK)         // 20480 halfs = 40960 B per weight k-chunk
#define CHUNK_BYTES (CHUNK_HALFS * 2)
#define NCHUNK (DD / 32)                // 16 k-chunks
#define WT_HALFS_PER_MAT (NCHUNK * CHUNK_HALFS)   // 327680

// ---- LDS layout (dynamic shared memory) ----
#define SX_OFF   0
#define SX_BYTES (64 * PADK * 2)                  // 5120
#define SW0_OFF  (SX_OFF + SX_BYTES)
#define SH_OFF   (SW0_OFF + 2 * CHUNK_BYTES)
#define PADH 520
#define SH_BYTES (64 * PADH * 2)                  // 66560
#define SMEM_BYTES (SH_OFF + SH_BYTES)            // 153600 B < 320KB WGP LDS

union V16U { v16bf v; v8bf h[2]; };

__device__ inline v16bf frag_ld(const bf16* p, int second_off) {
    V16U u;
    u.h[0] = *(const v8bf*)(p);
    u.h[1] = *(const v8bf*)(p + second_off);
    return u.v;
}

__device__ inline float gelu_exact(float x) {
    return 0.5f * x * (1.0f + erff(x * 0.70710678118654752f));
}

// CDNA5 async global->LDS copy, tracked with ASYNCcnt.
__device__ inline void async_ld_b128(unsigned lds_byte, const bf16* g) {
    asm volatile("global_load_async_to_lds_b128 %0, %1, off"
                 :: "v"(lds_byte), "v"((unsigned long long)(uintptr_t)g)
                 : "memory");
}
#define WAIT_ASYNC(n) asm volatile("s_wait_asynccnt " #n ::: "memory")

// Issue one 40KB weight k-chunk copy: 2560 x b128, 10 per thread (per-wave ASYNCcnt += 10).
__device__ inline void issue_chunk_copy(const bf16* wt_chunk, unsigned lds_base, int tid) {
#pragma unroll
    for (int j = 0; j < 10; ++j) {
        int i = tid + j * 256;                       // b128 index within chunk
        async_ld_b128(lds_base + (unsigned)i * 16u, wt_chunk + (size_t)i * 8);
    }
}

// Stage X chunk [64 rows x 32 k] f32 -> bf16 into sX (vectorized).
__device__ inline void stage_x(const float* __restrict__ X, int rowBase, int k0,
                               bf16* sX, int tid) {
#pragma unroll
    for (int j = 0; j < 2; ++j) {
        int i = tid + j * 256;                       // 512 float4s
        int r  = i >> 3;
        int ko = (i & 7) * 4;
        float4 f = *(const float4*)(X + (size_t)(rowBase + r) * DD + k0 + ko);
        v4bf h;
        h.x = (bf16)f.x; h.y = (bf16)f.y; h.z = (bf16)f.z; h.w = (bf16)f.w;
        *(v4bf*)(sX + r * PADK + ko) = h;            // 8B aligned
    }
}

// Y[rows,512] = GELU(X @ W1 + b1) @ W2 + b2, weights pre-staged as bf16 chunk-blocked.
__global__ void __launch_bounds__(256)
mlp_wmma_kernel(const float* __restrict__ X,
                const bf16* __restrict__ Wt1, const float* __restrict__ b1,
                const bf16* __restrict__ Wt2, const float* __restrict__ b2,
                float* __restrict__ Y)
{
    extern __shared__ char smem[];
    bf16* sX = (bf16*)(smem + SX_OFF);
    bf16* sH = (bf16*)(smem + SH_OFF);
    // LDS aperture uses addr[31:0]: low 32 bits of generic pointer = LDS byte offset
    const unsigned lds_base = (unsigned)(uintptr_t)smem;

    const int tid  = threadIdx.x;
    const int wave = tid >> 5;
    const int lane = tid & 31;
    const int l15  = lane & 15;
    const int hi   = lane >> 4;
    const int kbA  = hi * 8;        // A-fragment k base (16x32 bf16 layout)
    const int kbB  = hi * 8;        // B-fragment k base (same k order as A)
    const int n0   = wave * 64;
    const int rowBase = blockIdx.x * 64;

    v8f acc[4][4] = {};

    // ---------------- Stage A: H = GELU(X @ W1 + b1) ----------------
    issue_chunk_copy(Wt1, lds_base + SW0_OFF, tid);
    for (int c = 0; c < NCHUNK; ++c) {
        const unsigned cur = (unsigned)(c & 1);
        __syncthreads();                                    // prev compute done; buffers reusable
        stage_x(X, rowBase, c * 32, sX, tid);
        if (c + 1 < NCHUNK) {
            issue_chunk_copy(Wt1 + (size_t)(c + 1) * CHUNK_HALFS,
                             lds_base + SW0_OFF + (cur ^ 1u) * CHUNK_BYTES, tid);
            WAIT_ASYNC(10);                                 // chunk c landed (10 newer outstanding)
        } else {
            WAIT_ASYNC(0);
        }
        __syncthreads();                                    // X stores + async copies visible to all

        // current weight buffer as offset off the single smem base (keeps DS addrspace)
        const bf16* sWc = (const bf16*)(smem + SW0_OFF + cur * CHUNK_BYTES);

        v16bf a[4];
#pragma unroll
        for (int mi = 0; mi < 4; ++mi)
            a[mi] = frag_ld(sX + (mi * 16 + l15) * PADK + kbA, 16);
#pragma unroll
        for (int ni = 0; ni < 4; ++ni) {
            v16bf bfrag = frag_ld(sWc + (n0 + ni * 16 + l15) * PADK + kbB, 16);
#pragma unroll
            for (int mi = 0; mi < 4; ++mi)
                acc[mi][ni] = __builtin_amdgcn_wmma_f32_16x16x32_bf16(
                    false, a[mi], false, bfrag, (short)0, acc[mi][ni], false, false);
        }
    }
    __syncthreads();

    // epilogue A: bias + exact GELU -> sH (bf16)
#pragma unroll
    for (int ni = 0; ni < 4; ++ni) {
        int col = n0 + ni * 16 + l15;
        float bias = b1[col];
#pragma unroll
        for (int mi = 0; mi < 4; ++mi) {
#pragma unroll
            for (int r = 0; r < 8; ++r) {
                int row = mi * 16 + r + hi * 8;
                float x = acc[mi][ni][r] + bias;
                sH[row * PADH + col] = (bf16)gelu_exact(x);
            }
        }
    }
    __syncthreads();

    // ---------------- Stage B: Y = H @ W2 + b2 ----------------
    {
        v8f z = {};
#pragma unroll
        for (int mi = 0; mi < 4; ++mi)
#pragma unroll
            for (int ni = 0; ni < 4; ++ni)
                acc[mi][ni] = z;
    }

    issue_chunk_copy(Wt2, lds_base + SW0_OFF, tid);
    for (int c = 0; c < NCHUNK; ++c) {
        const unsigned cur = (unsigned)(c & 1);
        __syncthreads();
        if (c + 1 < NCHUNK) {
            issue_chunk_copy(Wt2 + (size_t)(c + 1) * CHUNK_HALFS,
                             lds_base + SW0_OFF + (cur ^ 1u) * CHUNK_BYTES, tid);
            WAIT_ASYNC(10);
        } else {
            WAIT_ASYNC(0);
        }
        __syncthreads();

        const bf16* sWc = (const bf16*)(smem + SW0_OFF + cur * CHUNK_BYTES);

        v16bf a[4];
#pragma unroll
        for (int mi = 0; mi < 4; ++mi)
            a[mi] = frag_ld(sH + (mi * 16 + l15) * PADH + c * 32 + kbA, 16);
#pragma unroll
        for (int ni = 0; ni < 4; ++ni) {
            v16bf bfrag = frag_ld(sWc + (n0 + ni * 16 + l15) * PADK + kbB, 16);
#pragma unroll
            for (int mi = 0; mi < 4; ++mi)
                acc[mi][ni] = __builtin_amdgcn_wmma_f32_16x16x32_bf16(
                    false, a[mi], false, bfrag, (short)0, acc[mi][ni], false, false);
        }
    }

    // epilogue B: bias, write f32 context (coalesced b32 per lane)
#pragma unroll
    for (int ni = 0; ni < 4; ++ni) {
        int col = n0 + ni * 16 + l15;
        float bias = b2[col];
#pragma unroll
        for (int mi = 0; mi < 4; ++mi) {
#pragma unroll
            for (int r = 0; r < 8; ++r) {
                int row = rowBase + mi * 16 + r + hi * 8;
                Y[(size_t)row * DD + col] = acc[mi][ni][r] + bias;
            }
        }
    }
}

// One-time: W[k,n] f32 -> Wt[c][n][40] bf16 (transposed, chunk-blocked, padded).
__global__ void __launch_bounds__(256)
prep_weight_kernel(const float* __restrict__ W, bf16* __restrict__ Wt)
{
    int t = blockIdx.x * blockDim.x + threadIdx.x;   // 0 .. 16*512-1
    int c = t >> 9;
    int n = t & (DD - 1);
    const float* src = W + (size_t)c * 32 * DD + n;
    bf16* dst = Wt + (size_t)c * CHUNK_HALFS + (size_t)n * PADK;
#pragma unroll
    for (int k = 0; k < 32; ++k) dst[k] = (bf16)src[(size_t)k * DD];
#pragma unroll
    for (int k = 32; k < PADK; ++k) dst[k] = (bf16)0.f;
}

__device__ inline float wave_sum32(float v) {
#pragma unroll
    for (int ofs = 16; ofs > 0; ofs >>= 1) v += __shfl_xor(v, ofs, 32);
    return v;
}

// jamo_out = LN(jamo + gather(ctxS, idx)); one wave per row, float4 vectorized.
__global__ void __launch_bounds__(256)
gather_ln_kernel(const float* __restrict__ jamo, const float* __restrict__ ctxS,
                 const int* __restrict__ sidx,
                 const float* __restrict__ g, const float* __restrict__ be,
                 float* __restrict__ out)
{
    int row = blockIdx.x * 8 + (threadIdx.x >> 5);
    if (row >= BATCH * SJN) return;
    int lane = threadIdx.x & 31;
    int b = row / SJN;
    int idx = sidx[row] - 1;
    bool valid = (idx >= 0) && (idx < SSN);
    const float4* jr = (const float4*)(jamo + (size_t)row * DD);
    const float4* cr = (const float4*)(ctxS + ((size_t)b * SSN + (valid ? idx : 0)) * DD);

    float4 v[4];
    float s = 0.f;
#pragma unroll
    for (int i = 0; i < 4; ++i) {
        float4 x = jr[lane + i * 32];
        if (valid) {
            float4 c4 = cr[lane + i * 32];
            x.x += c4.x; x.y += c4.y; x.z += c4.z; x.w += c4.w;
        }
        v[i] = x; s += x.x + x.y + x.z + x.w;
    }
    s = wave_sum32(s);
    float mean = s * (1.0f / DD);
    float vs = 0.f;
#pragma unroll
    for (int i = 0; i < 4; ++i) {
        float dx = v[i].x - mean, dy = v[i].y - mean, dz = v[i].z - mean, dw = v[i].w - mean;
        vs += dx * dx + dy * dy + dz * dz + dw * dw;
    }
    vs = wave_sum32(vs);
    float rstd = rsqrtf(vs * (1.0f / DD) + 1e-5f);
    float4* orow = (float4*)(out + (size_t)row * DD);
    const float4* g4 = (const float4*)g;
    const float4* b4 = (const float4*)be;
#pragma unroll
    for (int i = 0; i < 4; ++i) {
        int c4i = lane + i * 32;
        float4 gg = g4[c4i], bb = b4[c4i], o;
        o.x = (v[i].x - mean) * rstd * gg.x + bb.x;
        o.y = (v[i].y - mean) * rstd * gg.y + bb.y;
        o.z = (v[i].z - mean) * rstd * gg.z + bb.z;
        o.w = (v[i].w - mean) * rstd * gg.w + bb.w;
        orow[c4i] = o;
    }
}

// scatter-add jamo_context into per-syllable sums + counts (f32 atomics)
__global__ void __launch_bounds__(256)
scatter_add_kernel(const float* __restrict__ ctxJ, const int* __restrict__ sidx,
                   float* __restrict__ segsum, float* __restrict__ cnt)
{
    int row = blockIdx.x * 8 + (threadIdx.x >> 5);
    if (row >= BATCH * SJN) return;
    int lane = threadIdx.x & 31;
    int b = row / SJN;
    int idx = sidx[row] - 1;
    if (idx < 0 || idx >= SSN) return;   // wave-uniform branch (one row per wave)
    float* dst = segsum + ((size_t)b * SSN + idx) * DD;
    const float* src = ctxJ + (size_t)row * DD;
#pragma unroll
    for (int i = 0; i < 16; ++i) {
        int c = lane + i * 32;
        atomicAdd(dst + c, src[c]);
    }
    if (lane == 0) atomicAdd(cnt + b * SSN + idx, 1.0f);
}

// syl_out = LN(syl + segsum / max(cnt,1)); one wave per row, float4 vectorized.
__global__ void __launch_bounds__(256)
syl_ln_kernel(const float* __restrict__ syl, const float* __restrict__ segsum,
              const float* __restrict__ cnt,
              const float* __restrict__ g, const float* __restrict__ be,
              float* __restrict__ out)
{
    int row = blockIdx.x * 8 + (threadIdx.x >> 5);
    if (row >= BATCH * SSN) return;
    int lane = threadIdx.x & 31;
    float sc = 1.0f / fmaxf(cnt[row], 1.0f);
    const float4* sr = (const float4*)(syl + (size_t)row * DD);
    const float4* mr = (const float4*)(segsum + (size_t)row * DD);

    float4 v[4];
    float s = 0.f;
#pragma unroll
    for (int i = 0; i < 4; ++i) {
        float4 x = sr[lane + i * 32];
        float4 m4 = mr[lane + i * 32];
        x.x += m4.x * sc; x.y += m4.y * sc; x.z += m4.z * sc; x.w += m4.w * sc;
        v[i] = x; s += x.x + x.y + x.z + x.w;
    }
    s = wave_sum32(s);
    float mean = s * (1.0f / DD);
    float vs = 0.f;
#pragma unroll
    for (int i = 0; i < 4; ++i) {
        float dx = v[i].x - mean, dy = v[i].y - mean, dz = v[i].z - mean, dw = v[i].w - mean;
        vs += dx * dx + dy * dy + dz * dz + dw * dw;
    }
    vs = wave_sum32(vs);
    float rstd = rsqrtf(vs * (1.0f / DD) + 1e-5f);
    float4* orow = (float4*)(out + (size_t)row * DD);
    const float4* g4 = (const float4*)g;
    const float4* b4 = (const float4*)be;
#pragma unroll
    for (int i = 0; i < 4; ++i) {
        int c4i = lane + i * 32;
        float4 gg = g4[c4i], bb = b4[c4i], o;
        o.x = (v[i].x - mean) * rstd * gg.x + bb.x;
        o.y = (v[i].y - mean) * rstd * gg.y + bb.y;
        o.z = (v[i].z - mean) * rstd * gg.z + bb.z;
        o.w = (v[i].w - mean) * rstd * gg.w + bb.w;
        orow[c4i] = o;
    }
}

__global__ void zero_kernel(float* __restrict__ p, size_t n) {
    size_t stride = (size_t)gridDim.x * blockDim.x;
    for (size_t i = (size_t)blockIdx.x * blockDim.x + threadIdx.x; i < n; i += stride)
        p[i] = 0.0f;
}

extern "C" void kernel_launch(void* const* d_in, const int* in_sizes, int n_in,
                              void* d_out, int out_size, void* d_ws, size_t ws_size,
                              hipStream_t stream) {
    const float* jamo   = (const float*)d_in[0];
    const float* syl    = (const float*)d_in[1];
    const int*   sidx   = (const int*)d_in[2];
    const float* s2jW1  = (const float*)d_in[3];
    const float* s2jb1  = (const float*)d_in[4];
    const float* s2jW2  = (const float*)d_in[5];
    const float* s2jb2  = (const float*)d_in[6];
    const float* j2sW1  = (const float*)d_in[7];
    const float* j2sb1  = (const float*)d_in[8];
    const float* j2sW2  = (const float*)d_in[9];
    const float* j2sb2  = (const float*)d_in[10];
    const float* ln1g   = (const float*)d_in[11];
    const float* ln1b   = (const float*)d_in[12];
    const float* ln2g   = (const float*)d_in[13];
    const float* ln2b   = (const float*)d_in[14];

    float* out = (float*)d_out;
    float* ws  = (float*)d_ws;

    float* ctxS = ws;                                   // [B*SS*D] f32
    float* ctxJ = ctxS + (size_t)BATCH * SSN * DD;      // [B*SJ*D] f32
    float* segS = ctxJ + (size_t)BATCH * SJN * DD;      // [B*SS*D] f32
    float* cnt  = segS + (size_t)BATCH * SSN * DD;      // [B*SS]   f32
    bf16*  wtS1 = (bf16*)(cnt + (size_t)BATCH * SSN);   // 4 x 327680 bf16
    bf16*  wtS2 = wtS1 + WT_HALFS_PER_MAT;
    bf16*  wtJ1 = wtS2 + WT_HALFS_PER_MAT;
    bf16*  wtJ2 = wtJ1 + WT_HALFS_PER_MAT;

    size_t nz = (size_t)BATCH * SSN * DD + (size_t)BATCH * SSN;
    zero_kernel<<<1024, 256, 0, stream>>>(segS, nz);

    // one-time weight transpose/convert into chunk-blocked bf16
    prep_weight_kernel<<<32, 256, 0, stream>>>(s2jW1, wtS1);
    prep_weight_kernel<<<32, 256, 0, stream>>>(s2jW2, wtS2);
    prep_weight_kernel<<<32, 256, 0, stream>>>(j2sW1, wtJ1);
    prep_weight_kernel<<<32, 256, 0, stream>>>(j2sW2, wtJ2);

    mlp_wmma_kernel<<<(BATCH * SSN) / 64, 256, SMEM_BYTES, stream>>>(
        syl, wtS1, s2jb1, wtS2, s2jb2, ctxS);
    mlp_wmma_kernel<<<(BATCH * SJN) / 64, 256, SMEM_BYTES, stream>>>(
        jamo, wtJ1, j2sb1, wtJ2, j2sb2, ctxJ);

    gather_ln_kernel<<<(BATCH * SJN) / 8, 256, 0, stream>>>(
        jamo, ctxS, sidx, ln1g, ln1b, out);

    scatter_add_kernel<<<(BATCH * SJN) / 8, 256, 0, stream>>>(
        ctxJ, sidx, segS, cnt);

    syl_ln_kernel<<<(BATCH * SSN) / 8, 256, 0, stream>>>(
        syl, segS, cnt, ln2g, ln2b, out + (size_t)BATCH * SJN * DD);
}
